// EmbedMatcher_26645977104891
// MI455X (gfx1250) — compile-verified, matrix operand order checked by reading, not run on hardware
//
#include <hip/hip_runtime.h>
#include <cstdint>

#define PAD_IDX 200000
#define NBB 200

typedef __attribute__((ext_vector_type(16))) __bf16 bf16x16;
typedef __attribute__((ext_vector_type(8)))  __bf16 bf16x8;
typedef __attribute__((ext_vector_type(8)))  float  f32x8;
typedef __attribute__((ext_vector_type(4)))  float  f32x4;

__device__ inline f32x8 zero8() {
  f32x8 z;
#pragma unroll
  for (int i = 0; i < 8; ++i) z[i] = 0.f;
  return z;
}

__device__ inline f32x8 wmma_bf16(bf16x16 a, bf16x16 b, f32x8 c) {
  return __builtin_amdgcn_wmma_f32_16x16x32_bf16(false, a, false, b, (short)0, c, false, false);
}

// Fragment element e (0..7) holds K = kb+e ; e (8..15) holds K = kb+16+(e-8),
// kb = 0 for lanes 0-15, 8 for lanes 16-31. Row = base + (lane&15).
__device__ inline bf16x16 load_frag_bf16(const __bf16* __restrict__ W, int ld,
                                         int nbase, int kbase, int lane) {
  int r  = nbase + (lane & 15);
  int kb = kbase + ((lane & 16) ? 8 : 0);
  const __bf16* p = W + (size_t)r * ld + kb;
  bf16x8 lo = *(const bf16x8*)(p);
  bf16x8 hi = *(const bf16x8*)(p + 16);
  bf16x16 v;
#pragma unroll
  for (int i = 0; i < 8; ++i) { v[i] = lo[i]; v[i + 8] = hi[i]; }
  return v;
}

// Load the 8 K-slice fragments of one 16-column tile of W (row-major, ld elems).
__device__ __forceinline__ void load_btile(bf16x16* dst, const __bf16* __restrict__ W,
                                           int ld, int nbase, int lane) {
#pragma unroll
  for (int es = 0; es < 8; ++es)
    dst[es] = load_frag_bf16(W, ld, nbase, es * 32, lane);
}

__device__ __forceinline__ f32x8 wmma_chain8(const bf16x16* a, const bf16x16* b) {
  f32x8 acc = zero8();
#pragma unroll
  for (int es = 0; es < 8; ++es) acc = wmma_bf16(a[es], b[es], acc);
  return acc;
}

__device__ inline bf16x16 cvt_frag(f32x4 a, f32x4 b, f32x4 c, f32x4 d) {
  bf16x16 v;
#pragma unroll
  for (int i = 0; i < 4; ++i) {
    v[i]      = (__bf16)a[i];
    v[4 + i]  = (__bf16)b[i];
    v[8 + i]  = (__bf16)c[i];
    v[12 + i] = (__bf16)d[i];
  }
  return v;
}

__device__ inline float sigm(float x) { return 1.f / (1.f + __expf(-x)); }

// ---------------------------------------------------------------------------
// Prep: convert weights to bf16, fuse biases, zero padding rows of xall.
// ---------------------------------------------------------------------------
__global__ void k_prep(const float* __restrict__ gW, const float* __restrict__ p1W,
                       const float* __restrict__ p2W, const float* __restrict__ Wih,
                       const float* __restrict__ Whh, const float* __restrict__ bih,
                       const float* __restrict__ bhh,
                       __bf16* gWb, __bf16* W1b, __bf16* W2b, __bf16* Wihb,
                       __bf16* Wh1b, float* bsum, float* xall_f, __bf16* xall_b) {
  long i = (long)blockIdx.x * 256 + threadIdx.x;
  if (i < 32768)  { gWb[i]  = (__bf16)gW[i];  return; }  i -= 32768;
  if (i < 131072) { W1b[i]  = (__bf16)p1W[i]; return; }  i -= 131072;
  if (i < 131072) { W2b[i]  = (__bf16)p2W[i]; return; }  i -= 131072;
  if (i < 524288) { Wihb[i] = (__bf16)Wih[i]; return; }  i -= 524288;
  if (i < 524288) {
    long r = i >> 8, c = i & 255;
    Wh1b[i] = (__bf16)Whh[r * 512 + c];
    return;
  }
  i -= 524288;
  if (i < 2048) { bsum[i] = bih[i] + bhh[i]; return; }   i -= 2048;
  if (i < 11 * 256) {  // zero pad rows 1029..1039 of xall
    size_t o = (size_t)1029 * 256 + i;
    xall_f[o] = 0.f;
    xall_b[o] = (__bf16)0.f;
  }
}

// ---------------------------------------------------------------------------
// Neighbor GCN encoder: one block (8 waves) per sample.
// A fragments gathered straight from the embedding table into registers.
// dt loop ping-pong double-buffered (unrolled x2): B fragments for dt+1 are
// in flight while the dt WMMA chain executes -> loads overlap the matrix pipe.
// ---------------------------------------------------------------------------
__global__ __launch_bounds__(256) void k_neighbor(
    const int* __restrict__ conn, int outRowBase, int colOff,
    const float* __restrict__ emb, const __bf16* __restrict__ Wg,
    const float* __restrict__ wb, const float* __restrict__ bb,
    float* __restrict__ xall_f, __bf16* __restrict__ xall_b) {
  __shared__ float wmax[8][128];
  const int n = blockIdx.x;
  const int tid = threadIdx.x;
  const int wave = tid >> 5, lane = tid & 31;
  const int coll = lane & 15;
  const int hb = (lane & 16) ? 8 : 0;
  const int kb = (lane & 16) ? 8 : 0;

  if (lane < 16) {
#pragma unroll
    for (int dt = 0; dt < 8; ++dt) wmax[wave][dt * 16 + lane] = -3.0e38f;
  }

#pragma unroll 1
  for (int kt = wave; kt < 13; kt += 8) {
    if (kt + 8 < 13)
      __builtin_prefetch(&conn[((size_t)n * NBB + (kt + 8) * 16 + coll) * 2], 0, 1);

    const int row = kt * 16 + coll;
    int rel = PAD_IDX, ent = PAD_IDX;
    if (row < NBB) {
      rel = conn[((size_t)n * NBB + row) * 2 + 0];
      ent = conn[((size_t)n * NBB + row) * 2 + 1];
    }
    const bool bad = (row >= NBB) || (rel == PAD_IDX) || (ent == PAD_IDX);
    const unsigned int bmask = (unsigned int)__ballot(bad);  // bits 0..15 = rows
    const float* prel = emb + (size_t)rel * 128;
    const float* pent = emb + (size_t)ent * 128;

    bf16x16 afr[8];
#pragma unroll
    for (int es = 0; es < 8; ++es) {
      const float* base = (es < 4) ? (prel + es * 32) : (pent + (es - 4) * 32);
      const float* p = base + kb;
      afr[es] = cvt_frag(*(const f32x4*)(p),      *(const f32x4*)(p + 4),
                         *(const f32x4*)(p + 16), *(const f32x4*)(p + 20));
    }

    bf16x16 b0[8], b1[8];
    load_btile(b0, Wg, 256, 0, lane);           // dt = 0
#pragma unroll 1
    for (int dt = 0; dt < 8; dt += 2) {
      load_btile(b1, Wg, 256, (dt + 1) * 16, lane);   // in flight during b0 chain
      f32x8 acc = wmma_chain8(afr, b0);
      {
        const float bias = wb[dt * 16 + coll] + bb[dt * 16 + coll];
        float m = -3.0e38f;
#pragma unroll
        for (int r = 0; r < 8; ++r) {
          float v = acc[r] + bias;
          v = (v > 0.f) ? v : 0.1f * v;
          if ((bmask >> (hb + r)) & 1u) v = -3.0e38f;
          m = fmaxf(m, v);
        }
        m = fmaxf(m, __shfl_xor(m, 16, 32));
        if (lane < 16) {
          float* wp = &wmax[wave][dt * 16 + lane];
          *wp = fmaxf(*wp, m);
        }
      }
      if (dt + 2 < 8)
        load_btile(b0, Wg, 256, (dt + 2) * 16, lane); // in flight during b1 chain
      acc = wmma_chain8(afr, b1);
      {
        const float bias = wb[(dt + 1) * 16 + coll] + bb[(dt + 1) * 16 + coll];
        float m = -3.0e38f;
#pragma unroll
        for (int r = 0; r < 8; ++r) {
          float v = acc[r] + bias;
          v = (v > 0.f) ? v : 0.1f * v;
          if ((bmask >> (hb + r)) & 1u) v = -3.0e38f;
          m = fmaxf(m, v);
        }
        m = fmaxf(m, __shfl_xor(m, 16, 32));
        if (lane < 16) {
          float* wp = &wmax[wave][(dt + 1) * 16 + lane];
          *wp = fmaxf(*wp, m);
        }
      }
    }
  }
  __syncthreads();
  if (tid < 128) {
    float m = wmax[0][tid];
#pragma unroll
    for (int w = 1; w < 8; ++w) m = fmaxf(m, wmax[w][tid]);
    const float t = tanhf(m);
    const size_t orow = (size_t)(outRowBase + n);
    xall_f[orow * 256 + colOff + tid] = t;
    xall_b[orow * 256 + colOff + tid] = (__bf16)t;
  }
}

// ---------------------------------------------------------------------------
// Support/query encoder: relu(x@W1^T+b1)@W2^T + b2 + x, then LayerNorm.
// One wave per 16-row tile; hidden tile staged in LDS (bf16, pitch 520).
// ---------------------------------------------------------------------------
__global__ __launch_bounds__(32) void k_encoder(
    const float* __restrict__ xall_f, const __bf16* __restrict__ xall_b,
    const __bf16* __restrict__ W1b, const float* __restrict__ p1b,
    const __bf16* __restrict__ W2b, const float* __restrict__ p2b,
    const float* __restrict__ lng, const float* __restrict__ lnb,
    float* __restrict__ enc_f, __bf16* __restrict__ qg_b) {
  __shared__ __bf16 hbuf[16 * 520];
  __shared__ float  obuf[16 * 264];
  const int lane = threadIdx.x;
  const int rowbase = blockIdx.x * 16;
  const int coll = lane & 15;
  const int hb = (lane & 16) ? 8 : 0;

  bf16x16 afr[8];
#pragma unroll
  for (int es = 0; es < 8; ++es)
    afr[es] = load_frag_bf16(xall_b, 256, rowbase, es * 32, lane);

  // pass 1: h = relu(x @ W1^T + b1) -> LDS (bf16)
#pragma unroll 1
  for (int dt = 0; dt < 32; ++dt) {
    bf16x16 bfr[8];
    load_btile(bfr, W1b, 256, dt * 16, lane);
    f32x8 acc = wmma_chain8(afr, bfr);
    const float bias = p1b[dt * 16 + coll];
#pragma unroll
    for (int r = 0; r < 8; ++r) {
      float v = acc[r] + bias;
      v = (v > 0.f) ? v : 0.f;
      hbuf[(hb + r) * 520 + dt * 16 + coll] = (__bf16)v;
    }
  }
  __syncthreads();

  // pass 2: o = h @ W2^T + b2 + x -> LDS (f32), two K halves
#pragma unroll 1
  for (int half = 0; half < 2; ++half) {
    bf16x16 afr2[8];
#pragma unroll
    for (int e = 0; e < 8; ++e)
      afr2[e] = load_frag_bf16((const __bf16*)hbuf, 520, 0, (half * 8 + e) * 32, lane);
#pragma unroll 1
    for (int dt = 0; dt < 16; ++dt) {
      bf16x16 bfr[8];
#pragma unroll
      for (int e = 0; e < 8; ++e)
        bfr[e] = load_frag_bf16(W2b, 512, dt * 16, (half * 8 + e) * 32, lane);
      f32x8 acc = wmma_chain8(afr2, bfr);
      const float bias = p2b[dt * 16 + coll];
#pragma unroll
      for (int r = 0; r < 8; ++r) {
        const int rr = hb + r, cc = dt * 16 + coll;
        float v = acc[r];
        if (half == 0)
          v += bias + xall_f[(size_t)(rowbase + rr) * 256 + cc];
        else
          v += obuf[rr * 264 + cc];
        obuf[rr * 264 + cc] = v;
      }
    }
    __syncthreads();
  }

  // LayerNorm: lane pair (l, l^16) handles one row, 128 cols each half.
  const int rr = lane & 15;
  const int ch = (lane & 16) ? 128 : 0;
  float s = 0.f, ss = 0.f;
  for (int c = 0; c < 128; ++c) {
    const float v = obuf[rr * 264 + ch + c];
    s += v; ss += v * v;
  }
  s  += __shfl_xor(s, 16, 32);
  ss += __shfl_xor(ss, 16, 32);
  const float mean = s * (1.f / 256.f);
  const float var  = ss * (1.f / 256.f) - mean * mean;
  const float rstd = rsqrtf(var + 1e-5f);
  const int grow = rowbase + rr;
  for (int c = 0; c < 128; ++c) {
    const int cc = ch + c;
    const float v = (obuf[rr * 264 + cc] - mean) * rstd * lng[cc] + lnb[cc];
    enc_f[(size_t)grow * 256 + cc] = v;
    if (grow < 1024) qg_b[(size_t)grow * 256 + cc] = (__bf16)v;
  }
}

// ---------------------------------------------------------------------------
// support_g = mean of enc rows 1024..1028 ; hvec = W_hh[:,256:] @ support_g
// ---------------------------------------------------------------------------
__global__ __launch_bounds__(256) void k_prep2(const float* __restrict__ enc_f,
                                               const float* __restrict__ Whh,
                                               float* __restrict__ sg,
                                               float* __restrict__ hvec) {
  __shared__ float ssg[256];
  const int t = threadIdx.x;
  float m = 0.f;
#pragma unroll
  for (int r = 0; r < 5; ++r) m += enc_f[(size_t)(1024 + r) * 256 + t];
  m *= 0.2f;
  sg[t] = m;
  ssg[t] = m;
  __syncthreads();
  for (int j = t; j < 2048; j += 256) {
    const float* wr = Whh + (size_t)j * 512 + 256;
    float a = 0.f;
    for (int c = 0; c < 256; ++c) a += wr[c] * ssg[c];
    hvec[j] = a;
  }
}

// ---------------------------------------------------------------------------
// qgate = query_g @ W_ih^T + (b_ih + b_hh)   (step-invariant, computed once)
// dt loop ping-pong double-buffered.
// ---------------------------------------------------------------------------
__global__ __launch_bounds__(128) void k_qgate(const __bf16* __restrict__ qg_b,
                                               const __bf16* __restrict__ Wihb,
                                               const float* __restrict__ bsum,
                                               float* __restrict__ qgate) {
  const int rowbase = blockIdx.x * 16;
  const int wave = threadIdx.x >> 5, lane = threadIdx.x & 31;
  const int coll = lane & 15;
  const int hb = (lane & 16) ? 8 : 0;
  bf16x16 afr[8];
#pragma unroll
  for (int es = 0; es < 8; ++es)
    afr[es] = load_frag_bf16(qg_b, 256, rowbase, es * 32, lane);

  const int dt0 = wave * 32;
  bf16x16 b0[8], b1[8];
  load_btile(b0, Wihb, 256, dt0 * 16, lane);
#pragma unroll 1
  for (int dtl = 0; dtl < 32; dtl += 2) {
    const int dt = dt0 + dtl;
    load_btile(b1, Wihb, 256, (dt + 1) * 16, lane);
    f32x8 acc = wmma_chain8(afr, b0);
    {
      const int cc = dt * 16 + coll;
      const float bs = bsum[cc];
#pragma unroll
      for (int r = 0; r < 8; ++r)
        qgate[(size_t)(rowbase + hb + r) * 2048 + cc] = acc[r] + bs;
    }
    if (dtl + 2 < 32) load_btile(b0, Wihb, 256, (dt + 2) * 16, lane);
    acc = wmma_chain8(afr, b1);
    {
      const int cc = (dt + 1) * 16 + coll;
      const float bs = bsum[cc];
#pragma unroll
      for (int r = 0; r < 8; ++r)
        qgate[(size_t)(rowbase + hb + r) * 2048 + cc] = acc[r] + bs;
    }
  }
}

// ---------------------------------------------------------------------------
// gates = h @ W_hh[:, :256]^T + qgate + hvec   (steps 1..3)
// dt loop ping-pong double-buffered.
// ---------------------------------------------------------------------------
__global__ __launch_bounds__(128) void k_gates(const __bf16* __restrict__ h_b,
                                               const __bf16* __restrict__ Wh1b,
                                               const float* __restrict__ qgate,
                                               const float* __restrict__ hvec,
                                               float* __restrict__ gates) {
  const int rowbase = blockIdx.x * 16;
  const int wave = threadIdx.x >> 5, lane = threadIdx.x & 31;
  const int coll = lane & 15;
  const int hb = (lane & 16) ? 8 : 0;
  bf16x16 afr[8];
#pragma unroll
  for (int es = 0; es < 8; ++es)
    afr[es] = load_frag_bf16(h_b, 256, rowbase, es * 32, lane);

  const int dt0 = wave * 32;
  bf16x16 b0[8], b1[8];
  load_btile(b0, Wh1b, 256, dt0 * 16, lane);
#pragma unroll 1
  for (int dtl = 0; dtl < 32; dtl += 2) {
    const int dt = dt0 + dtl;
    load_btile(b1, Wh1b, 256, (dt + 1) * 16, lane);
    f32x8 acc = wmma_chain8(afr, b0);
    {
      const int cc = dt * 16 + coll;
      const float hv = hvec[cc];
#pragma unroll
      for (int r = 0; r < 8; ++r) {
        const size_t o = (size_t)(rowbase + hb + r) * 2048 + cc;
        gates[o] = acc[r] + qgate[o] + hv;
      }
    }
    if (dtl + 2 < 32) load_btile(b0, Wh1b, 256, (dt + 2) * 16, lane);
    acc = wmma_chain8(afr, b1);
    {
      const int cc = (dt + 1) * 16 + coll;
      const float hv = hvec[cc];
#pragma unroll
      for (int r = 0; r < 8; ++r) {
        const size_t o = (size_t)(rowbase + hb + r) * 2048 + cc;
        gates[o] = acc[r] + qgate[o] + hv;
      }
    }
  }
}

// ---------------------------------------------------------------------------
// LSTM cell elementwise update. step==0 ignores prior c (treated as zero).
// ---------------------------------------------------------------------------
__global__ __launch_bounds__(256) void k_update(const float* __restrict__ gates,
                                                const float* __restrict__ qg_f,
                                                float* __restrict__ c_state,
                                                float* __restrict__ h_f,
                                                __bf16* __restrict__ h_b, int step) {
  const int idx = blockIdx.x * 256 + threadIdx.x;   // < 1024*512
  const int row = idx >> 9, cc = idx & 511;
  const size_t base = (size_t)row * 2048;
  const float gi = gates[base + cc];
  const float gf = gates[base + 512 + cc];
  const float gg = gates[base + 1024 + cc];
  const float go = gates[base + 1536 + cc];
  const float cprev = step ? c_state[idx] : 0.f;
  const float c = sigm(gf) * cprev + sigm(gi) * tanhf(gg);
  c_state[idx] = c;
  const float hn = sigm(go) * tanhf(c);
  if (cc < 256) {
    const float hv = qg_f[(size_t)row * 256 + cc] + hn;
    h_f[(size_t)row * 256 + cc] = hv;
    h_b[(size_t)row * 256 + cc] = (__bf16)hv;
  }
}

// ---------------------------------------------------------------------------
// out[n] = dot(h[n], support_g)
// ---------------------------------------------------------------------------
__global__ __launch_bounds__(256) void k_out(const float* __restrict__ h_f,
                                             const float* __restrict__ sg,
                                             float* __restrict__ out) {
  const int wave = threadIdx.x >> 5, lane = threadIdx.x & 31;
  const int row = blockIdx.x * 8 + wave;
  const float* hr = h_f + (size_t)row * 256;
  float a = 0.f;
#pragma unroll
  for (int c = lane; c < 256; c += 32) a += hr[c] * sg[c];
#pragma unroll
  for (int off = 16; off >= 1; off >>= 1) a += __shfl_xor(a, off, 32);
  if (lane == 0) out[row] = a;
}

// ---------------------------------------------------------------------------
extern "C" void kernel_launch(void* const* d_in, const int* in_sizes, int n_in,
                              void* d_out, int out_size, void* d_ws, size_t ws_size,
                              hipStream_t stream) {
  const int* qlc = (const int*)d_in[0];
  const int* qrc = (const int*)d_in[1];
  const int* slc = (const int*)d_in[2];
  const int* src = (const int*)d_in[3];
  const float* emb = (const float*)d_in[8];
  const float* gW  = (const float*)d_in[9];
  const float* gwb = (const float*)d_in[10];
  const float* gb  = (const float*)d_in[11];
  const float* p1W = (const float*)d_in[12];
  const float* p1b = (const float*)d_in[13];
  const float* p2W = (const float*)d_in[14];
  const float* p2b = (const float*)d_in[15];
  const float* lng = (const float*)d_in[16];
  const float* lnb = (const float*)d_in[17];
  const float* Wih = (const float*)d_in[18];
  const float* Whh = (const float*)d_in[19];
  const float* bih = (const float*)d_in[20];
  const float* bhh = (const float*)d_in[21];
  float* out = (float*)d_out;

  char* w = (char*)d_ws;
  auto alloc = [&](size_t bytes) -> char* {
    char* p = w;
    w += (bytes + 255) & ~(size_t)255;
    return p;
  };
  __bf16* gWb    = (__bf16*)alloc(32768 * 2);
  __bf16* W1b    = (__bf16*)alloc(131072 * 2);
  __bf16* W2b    = (__bf16*)alloc(131072 * 2);
  __bf16* Wihb   = (__bf16*)alloc(524288 * 2);
  __bf16* Wh1b   = (__bf16*)alloc(524288 * 2);
  float*  xall_f = (float*) alloc((size_t)1040 * 256 * 4);
  __bf16* xall_b = (__bf16*)alloc((size_t)1040 * 256 * 2);
  float*  enc_f  = (float*) alloc((size_t)1040 * 256 * 4);
  __bf16* qg_b   = (__bf16*)alloc((size_t)1024 * 256 * 2);
  float*  sg     = (float*) alloc(256 * 4);
  float*  hvec   = (float*) alloc(2048 * 4);
  float*  bsum   = (float*) alloc(2048 * 4);
  float*  qgate  = (float*) alloc((size_t)1024 * 2048 * 4);
  float*  gates  = (float*) alloc((size_t)1024 * 2048 * 4);
  float*  cstate = (float*) alloc((size_t)1024 * 512 * 4);
  float*  h_f    = (float*) alloc((size_t)1024 * 256 * 4);
  __bf16* h_b    = (__bf16*)alloc((size_t)1024 * 256 * 2);

  const long prepTotal = 32768 + 131072 + 131072 + 524288 + 524288 + 2048 + 11 * 256;
  k_prep<<<(unsigned)((prepTotal + 255) / 256), 256, 0, stream>>>(
      gW, p1W, p2W, Wih, Whh, bih, bhh, gWb, W1b, W2b, Wihb, Wh1b, bsum, xall_f, xall_b);

  k_neighbor<<<1024, 256, 0, stream>>>(qlc, 0, 0,   emb, gWb, gwb, gb, xall_f, xall_b);
  k_neighbor<<<1024, 256, 0, stream>>>(qrc, 0, 128, emb, gWb, gwb, gb, xall_f, xall_b);
  k_neighbor<<<5,    256, 0, stream>>>(slc, 1024, 0,   emb, gWb, gwb, gb, xall_f, xall_b);
  k_neighbor<<<5,    256, 0, stream>>>(src, 1024, 128, emb, gWb, gwb, gb, xall_f, xall_b);

  k_encoder<<<65, 32, 0, stream>>>(xall_f, xall_b, W1b, p1b, W2b, p2b, lng, lnb, enc_f, qg_b);
  k_prep2<<<1, 256, 0, stream>>>(enc_f, Whh, sg, hvec);
  k_qgate<<<64, 128, 0, stream>>>(qg_b, Wihb, bsum, qgate);

  for (int s = 0; s < 4; ++s) {
    const float* gsrc = qgate;
    if (s > 0) {
      k_gates<<<64, 128, 0, stream>>>(h_b, Wh1b, qgate, hvec, gates);
      gsrc = gates;
    }
    k_update<<<(1024 * 512) / 256, 256, 0, stream>>>(gsrc, enc_f, cstate, h_f, h_b, s);
  }
  k_out<<<128, 256, 0, stream>>>(h_f, sg, out);
}
